// AdapterSwinTransformerBlock_12850542149948
// MI455X (gfx1250) — matrix operand order repaired and checked by reading, not compiled, not run on hardware
//
#include <hip/hip_runtime.h>
#include <hip/hip_bf16.h>

typedef __attribute__((ext_vector_type(8)))  __bf16 v8bf;
typedef __attribute__((ext_vector_type(16))) __bf16 v16bf;
typedef __attribute__((ext_vector_type(8)))  float  v8f;

#define WMMA_BF16(a,b,c) __builtin_amdgcn_wmma_f32_16x16x32_bf16(false,(a),false,(b),(short)0,(c),false,false)

static __device__ __forceinline__ v16bf mk16(v8bf lo, v8bf hi) {
  v16bf r;
#pragma unroll
  for (int i = 0; i < 8; ++i) { r[i] = lo[i]; r[i + 8] = hi[i]; }
  return r;
}

// CDNA5 async global->LDS copy (ASYNCcnt-tracked, bypasses VGPRs).
// vdst = per-lane LDS byte address (low 32 bits of a generic LDS pointer,
// ISA 10.2); vaddr = per-lane 64-bit global address.
static __device__ __forceinline__ void async_ld_b128(unsigned lds, const void* g) {
  asm volatile("global_load_async_to_lds_b128 %0, %1, off" :: "v"(lds), "v"(g) : "memory");
}
static __device__ __forceinline__ void wait_async(void) {
  asm volatile("s_wait_asynccnt 0x0" ::: "memory");
}
static __device__ __forceinline__ void wait_async4(void) {
  asm volatile("s_wait_asynccnt 0x4" ::: "memory");
}

// ---------------------------------------------------------------------------
// f32 -> bf16 weight conversion (run once per launch; ~0.8M elements total,
// converted weights stay L2-resident: 192MB L2 >> 1.6MB of bf16 weights).
// ---------------------------------------------------------------------------
__global__ __launch_bounds__(256) void cvt_kernel(
    const float* __restrict__ in, __bf16* __restrict__ out, int n)
{
  int i = blockIdx.x * 256 + threadIdx.x;
  if (i < n) out[i] = (__bf16)in[i];
}

// ---------------------------------------------------------------------------
// LayerNorm. WINDOW=true fuses cyclic shift (-3,-3) + window partition as a
// row gather (rows permute whole, LN is per-row). One wave per 256-ch row.
// ---------------------------------------------------------------------------
template<bool WINDOW>
__global__ __launch_bounds__(256) void ln_kernel(
    const float* __restrict__ in, const float* __restrict__ g,
    const float* __restrict__ bta, __bf16* __restrict__ out)
{
  const int wave = threadIdx.x >> 5, lane = threadIdx.x & 31;
  const long m = (long)blockIdx.x * 8 + wave;
  const float* src;
  if (WINDOW) {
    int bb = (int)(m / 3136); int rem = (int)(m % 3136);
    int win = rem / 49, tok = rem % 49;
    int wh = win >> 3, ww = win & 7, th = tok / 7, tw = tok - th * 7;
    int h = wh * 7 + th + 3; if (h >= 56) h -= 56;
    int w = ww * 7 + tw + 3; if (w >= 56) w -= 56;
    src = in + ((long)bb * 3136 + h * 56 + w) * 256;
  } else {
    src = in + m * 256;
  }
  float v[8]; float s = 0.f, ss = 0.f;
#pragma unroll
  for (int k = 0; k < 8; ++k) {
    int c = lane + k * 32; float t = src[c];
    v[k] = t; s += t; ss += t * t;
  }
#pragma unroll
  for (int d = 1; d < 32; d <<= 1) { s += __shfl_xor(s, d, 32); ss += __shfl_xor(ss, d, 32); }
  float mu = s * (1.f / 256.f);
  float var = ss * (1.f / 256.f) - mu * mu;
  float rs = rsqrtf(var + 1e-5f);
#pragma unroll
  for (int k = 0; k < 8; ++k) {
    int c = lane + k * 32;
    out[m * 256 + c] = (__bf16)((v[k] - mu) * rs * g[c] + bta[c]);
  }
}

// ---------------------------------------------------------------------------
// bf16 WMMA GEMM:  C[M,N] = A[M,K] * W[N,K]^T  (+ fused epilogue).
// 128x128 block tile, K-step 32, 8 waves (2M x 4N), 4x2 16x16 tiles/wave.
// Double-buffered LDS fed by async global->LDS b128 copies: the copy for
// K-step i+2 overlaps the WMMA work of step i+1 (s_wait_asynccnt 0 for own
// loads, then workgroup barrier, then issue into the freed buffer).
// LDS rows padded to 40 bf16 (80B = 20 banks) -> conflict-free ds_read_b128.
// ---------------------------------------------------------------------------
enum { EPI_QKV = 0, EPI_PROJ = 1, EPI_FC1 = 2, EPI_FC2 = 3 };

template<int N_, int K_, int EPI>
__global__ __launch_bounds__(256) void gemm_kernel(
    const __bf16* __restrict__ A, const __bf16* __restrict__ W,
    const float* __restrict__ bias, const float* __restrict__ aux,
    __bf16* __restrict__ outb, float* __restrict__ outf)
{
  constexpr int NSTEP = K_ / 32;
  __shared__ __bf16 sA[2][128 * 40];
  __shared__ __bf16 sB[2][128 * 40];
  const int tid = threadIdx.x;
  const int wave = tid >> 5, lane = tid & 31;
  const int wm = wave & 1, wn = wave >> 1;
  const long m0 = (long)blockIdx.y * 128;
  const int  n0 = blockIdx.x * 128;

  const int lrow  = tid >> 1;        // 0..127 : tile row this thread stages
  const int lhalf = (tid & 1) << 4;  // 0 or 16 : K-half this thread stages

  const __bf16* agbase = A + (m0 + lrow) * K_ + lhalf;
  const __bf16* bgbase = W + (long)(n0 + lrow) * K_ + lhalf;

  auto issue = [&](int p, int k0) {
    unsigned la = (unsigned)(uintptr_t)&sA[p][lrow * 40 + lhalf];
    unsigned lb = (unsigned)(uintptr_t)&sB[p][lrow * 40 + lhalf];
    const __bf16* ga = agbase + k0;
    const __bf16* gb = bgbase + k0;
    async_ld_b128(la,      ga);
    async_ld_b128(la + 16, ga + 8);
    async_ld_b128(lb,      gb);
    async_ld_b128(lb + 16, gb + 8);
  };

  v8f acc[4][2] = {};
  // ISA fragment addressing (wave32):
  const int ar = lane & 15;          // A: row within 16-tile
  const int ak = (lane >> 4) << 3;   // A: K base 0/8 (+16 for hi chunk)
  const int bc = lane & 15;          // B: col within 16-tile
  const int bk = (lane >> 4) << 3;   // B: K base 0/8 (+16 for hi chunk)

  // pipeline prologue: buffers 0 and 1 in flight
  issue(0, 0);
  if constexpr (NSTEP > 1) { issue(1, 32); wait_async4(); }
  else                     { wait_async(); }
  __syncthreads();

  for (int i = 0; i < NSTEP; ++i) {
    const int p = i & 1;
    v16bf af[4];
#pragma unroll
    for (int t = 0; t < 4; ++t) {
      const __bf16* pp = &sA[p][(wm * 64 + t * 16 + ar) * 40 + ak];
      af[t] = mk16(*(const v8bf*)pp, *(const v8bf*)(pp + 16));
    }
#pragma unroll
    for (int j = 0; j < 2; ++j) {
      const __bf16* pp = &sB[p][(wn * 32 + j * 16 + bc) * 40 + bk];
      v16bf bfr = mk16(*(const v8bf*)pp, *(const v8bf*)(pp + 16));
#pragma unroll
      for (int t = 0; t < 4; ++t)
        acc[t][j] = WMMA_BF16(af[t], bfr, acc[t][j]);
    }
    if (i + 1 < NSTEP) {
      wait_async();            // my loads for step i+1 have landed in LDS
      __syncthreads();         // everyone's landed; everyone done with buf p
      if (i + 2 < NSTEP) issue(p, (i + 2) * 32);   // overlaps compute(i+1)
    }
  }

  // ---- fused epilogue; C layout: VGPR v = row (v or v+8), lane&15 = col ----
  const int rhalf = (lane >> 4) << 3;
#pragma unroll
  for (int i = 0; i < 4; ++i) {
#pragma unroll
    for (int j = 0; j < 2; ++j) {
      const int col = n0 + wn * 32 + j * 16 + (lane & 15);
      const float bs = bias[col];
#pragma unroll
      for (int v = 0; v < 8; ++v) {
        const long row = m0 + wm * 64 + i * 16 + rhalf + v;
        float val = acc[i][j][v] + bs;
        if constexpr (EPI == EPI_QKV) {
          outb[row * N_ + col] = (__bf16)val;
        } else if constexpr (EPI == EPI_FC1) {
          val = 0.5f * val * (1.f + erff(val * 0.70710678118654752f)); // exact GELU
          outb[row * N_ + col] = (__bf16)val;
        } else if constexpr (EPI == EPI_FC2) {
          const long idx = row * N_ + col;
          outf[idx] = val + aux[idx];                 // + x1 residual
        } else { // EPI_PROJ: window-reverse + roll(+3,+3) scatter + shortcut
          int bb = (int)(row / 3136); int rem = (int)(row % 3136);
          int win = rem / 49, tok = rem % 49;
          int wh = win >> 3, ww = win & 7, th = tok / 7, tw = tok - th * 7;
          int h = wh * 7 + th + 3; if (h >= 56) h -= 56;
          int w = ww * 7 + tw + 3; if (w >= 56) w -= 56;
          const long dst = ((long)bb * 3136 + h * 56 + w) * 256 + col;
          outf[dst] = val + aux[dst];                 // + shortcut x
        }
      }
    }
  }
}

// ---------------------------------------------------------------------------
// Windowed attention: one wave per (window, head). N=49 padded to 64.
// QK^T: head_dim=32 == WMMA K -> one v_wmma per 16x16 tile (16 total).
// Rel-pos bias and shift mask are closed-form (module constants in the
// reference) and computed in-register; column-side terms hoisted per j.
// Softmax reduces over 16-lane halves (C-fragment rows live per half).
// P and V^T re-fragmented through per-wave LDS for P*V (16 more WMMAs).
// Block = 4 waves (4 heads), 2 blocks per window -> 55KB LDS/block.
// ---------------------------------------------------------------------------
__global__ __launch_bounds__(128) void attn_kernel(
    const __bf16* __restrict__ qkv, const float* __restrict__ rel,
    __bf16* __restrict__ out)
{
  __shared__ __bf16 sP[4][64 * 72];   // P (post-softmax), padded rows
  __shared__ __bf16 sVT[4][32 * 72];  // V transposed: [channel][token]

  const int wv = threadIdx.x >> 5, lane = threadIdx.x & 31;
  const int head = ((blockIdx.x & 1) << 2) + wv;
  const int wing = blockIdx.x >> 1;        // global window id 0..4095
  const int w_in = wing & 63;              // window within batch
  const int wh = w_in >> 3, ww = w_in & 7;
  const long rowbase = (long)wing * 49;
  const int ldq = 768;

  const int ar = lane & 15, akk = (lane >> 4) << 3;
  const int bc = lane & 15, bkk = (lane >> 4) << 3;
  const int rhalf = (lane >> 4) << 3;

  // ---- Q fragments straight from global (row = token, K = 32 channels) ----
  v16bf qa[4];
#pragma unroll
  for (int i = 0; i < 4; ++i) {
    int tok = i * 16 + ar; if (tok > 48) tok = 48;     // pad rows clamped
    const __bf16* p = qkv + (rowbase + tok) * ldq + head * 32 + akk;
    qa[i] = mk16(*(const v8bf*)p, *(const v8bf*)(p + 16));
  }

  // ---- scores = Q K^T : 16 WMMAs ----
  v8f s[4][4] = {};
#pragma unroll
  for (int j = 0; j < 4; ++j) {
    int tok = j * 16 + bc; if (tok > 48) tok = 48;
    const __bf16* p = qkv + (rowbase + tok) * ldq + 256 + head * 32 + bkk;
    v16bf kb = mk16(*(const v8bf*)p, *(const v8bf*)(p + 16));
#pragma unroll
    for (int i = 0; i < 4; ++i)
      s[i][j] = WMMA_BF16(qa[i], kb, s[i][j]);
  }

  // ---- scale + rel-pos bias + shift mask (column terms hoisted) ----
  const float scale = 0.17677669529663689f;            // 32^-0.5
  int jh_[4], jw_[4], lj_[4]; bool cok[4];
#pragma unroll
  for (int j = 0; j < 4; ++j) {
    int col = j * 16 + (lane & 15);
    cok[j] = col < 49;
    int jh = col / 7, jw = col - jh * 7;
    jh_[j] = jh; jw_[j] = jw;
    int gh = wh * 7 + jh, gw = ww * 7 + jw;
    lj_[j] = (gh < 49 ? 0 : (gh < 53 ? 1 : 2)) * 3 + (gw < 49 ? 0 : (gw < 53 ? 1 : 2));
  }
#pragma unroll
  for (int i = 0; i < 4; ++i) {
#pragma unroll
    for (int v = 0; v < 8; ++v) {
      const int row = i * 16 + rhalf + v;
      const bool rok = row < 49;
      int ih = row / 7, iw = row - ih * 7;
      int gh = wh * 7 + ih, gw = ww * 7 + iw;
      int li = (gh < 49 ? 0 : (gh < 53 ? 1 : 2)) * 3 + (gw < 49 ? 0 : (gw < 53 ? 1 : 2));
#pragma unroll
      for (int j = 0; j < 4; ++j) {
        float val;
        if (rok && cok[j]) {
          int idx = (ih - jh_[j] + 6) * 13 + (iw - jw_[j] + 6);
          val = s[i][j][v] * scale + rel[idx * 8 + head] + (li != lj_[j] ? -100.0f : 0.0f);
        } else {
          val = -1e30f;                                 // pad -> exp = 0
        }
        s[i][j][v] = val;
      }
    }
  }

  // ---- softmax over cols (row lives across 16 lanes of one half) ----
#pragma unroll
  for (int i = 0; i < 4; ++i) {
#pragma unroll
    for (int v = 0; v < 8; ++v) {
      float mx = -3e38f;
#pragma unroll
      for (int j = 0; j < 4; ++j) mx = fmaxf(mx, s[i][j][v]);
#pragma unroll
      for (int d = 1; d < 16; d <<= 1) mx = fmaxf(mx, __shfl_xor(mx, d, 32));
      float sum = 0.f;
#pragma unroll
      for (int j = 0; j < 4; ++j) { float e = __expf(s[i][j][v] - mx); s[i][j][v] = e; sum += e; }
#pragma unroll
      for (int d = 1; d < 16; d <<= 1) sum += __shfl_xor(sum, d, 32);
      float inv = 1.0f / sum;
#pragma unroll
      for (int j = 0; j < 4; ++j) s[i][j][v] *= inv;
    }
  }

  // ---- stage P (bf16) and V^T into LDS for re-fragmentation ----
#pragma unroll
  for (int i = 0; i < 4; ++i)
#pragma unroll
    for (int j = 0; j < 4; ++j)
#pragma unroll
      for (int v = 0; v < 8; ++v)
        sP[wv][(i * 16 + rhalf + v) * 72 + j * 16 + (lane & 15)] = (__bf16)s[i][j][v];

  for (int t = lane; t < 64; t += 32) {
    const __bf16* vp = qkv + (rowbase + (t < 49 ? t : 48)) * ldq + 512 + head * 32;
#pragma unroll
    for (int c = 0; c < 32; ++c)
      sVT[wv][c * 72 + t] = (t < 49) ? vp[c] : (__bf16)0.0f;
  }
  __syncthreads();

  // ---- out = P * V : K=64 -> 2 wmma per tile; 4 token-tiles x 2 ch-tiles ----
#pragma unroll
  for (int i = 0; i < 4; ++i) {
    v8f o0 = {}, o1 = {};
#pragma unroll
    for (int kk = 0; kk < 2; ++kk) {
      const __bf16* pa = &sP[wv][(i * 16 + ar) * 72 + kk * 32 + akk];
      v16bf af = mk16(*(const v8bf*)pa, *(const v8bf*)(pa + 16));
      const __bf16* pb0 = &sVT[wv][bc * 72 + kk * 32 + bkk];
      v16bf vb0 = mk16(*(const v8bf*)pb0, *(const v8bf*)(pb0 + 16));
      const __bf16* pb1 = &sVT[wv][(16 + bc) * 72 + kk * 32 + bkk];
      v16bf vb1 = mk16(*(const v8bf*)pb1, *(const v8bf*)(pb1 + 16));
      o0 = WMMA_BF16(af, vb0, o0);
      o1 = WMMA_BF16(af, vb1, o1);
    }
#pragma unroll
    for (int v = 0; v < 8; ++v) {
      int tok = i * 16 + rhalf + v;
      if (tok < 49) {
        long r = (rowbase + tok) * 256 + head * 32;
        out[r + (lane & 15)]      = (__bf16)o0[v];
        out[r + 16 + (lane & 15)] = (__bf16)o1[v];
      }
    }
  }
}

// ---------------------------------------------------------------------------
extern "C" void kernel_launch(void* const* d_in, const int* in_sizes, int n_in,
                              void* d_out, int out_size, void* d_ws, size_t ws_size,
                              hipStream_t stream)
{
  (void)in_sizes; (void)n_in; (void)out_size; (void)ws_size;
  const float* x      = (const float*)d_in[0];
  const float* n1g    = (const float*)d_in[1];
  const float* n1b    = (const float*)d_in[2];
  const float* qkv_w  = (const float*)d_in[3];
  const float* qkv_b  = (const float*)d_in[4];
  const float* relt   = (const float*)d_in[5];
  const float* proj_w = (const float*)d_in[6];
  const float* proj_b = (const float*)d_in[7];
  const float* n2g    = (const float*)d_in[8];
  const float* n2b    = (const float*)d_in[9];
  const float* fc1_w  = (const float*)d_in[10];
  const float* fc1_b  = (const float*)d_in[11];
  const float* fc2_w  = (const float*)d_in[12];
  const float* fc2_b  = (const float*)d_in[13];

  const long M = 200704;                    // 64 * 3136 tokens
  char* ws = (char*)d_ws;
  // workspace layout (regions reused once dead):
  float*  x1   = (float*)ws;                                   // M*256 f32
  __bf16* xw   = (__bf16*)(ws + (size_t)M * 256 * 4);          // M*256 bf16 (LN1 out, reused as LN2 out)
  __bf16* attn = (__bf16*)((char*)xw + (size_t)M * 256 * 2);   // M*256 bf16
  __bf16* qkvb = (__bf16*)((char*)attn + (size_t)M * 256 * 2); // max(M*768,M*1024) bf16 (qkv, reused as fc1 out)
  char*   wtop = (char*)qkvb + (size_t)M * 1024 * 2;           // bf16 weight copies
  __bf16* qkvw_h = (__bf16*)wtop;                              // 768*256
  __bf16* projw_h = qkvw_h + 768 * 256;                        // 256*256
  __bf16* fc1w_h = projw_h + 256 * 256;                        // 1024*256
  __bf16* fc2w_h = fc1w_h + 1024 * 256;                        // 256*1024
  __bf16* h2   = xw;
  __bf16* fc1o = qkvb;

  // 0) one-time (per launch) weight conversion to bf16
  cvt_kernel<<<768,  256, 0, stream>>>(qkv_w,  qkvw_h, 768 * 256);
  cvt_kernel<<<256,  256, 0, stream>>>(proj_w, projw_h, 256 * 256);
  cvt_kernel<<<1024, 256, 0, stream>>>(fc1_w,  fc1w_h, 1024 * 256);
  cvt_kernel<<<1024, 256, 0, stream>>>(fc2_w,  fc2w_h, 256 * 1024);

  // 1) LN1 + shift + window partition -> bf16
  ln_kernel<true><<<25088, 256, 0, stream>>>(x, n1g, n1b, xw);
  // 2) QKV GEMM [M,256]x[768,256]^T + bias -> bf16
  gemm_kernel<768, 256, EPI_QKV><<<dim3(6, 1568), 256, 0, stream>>>(
      xw, qkvw_h, qkv_b, nullptr, qkvb, nullptr);
  // 3) windowed attention (QK^T, bias+mask, softmax, PV) -> bf16
  attn_kernel<<<8192, 128, 0, stream>>>(qkvb, relt, attn);
  // 4) proj GEMM + bias + window-reverse/roll scatter + shortcut -> f32 x1
  gemm_kernel<256, 256, EPI_PROJ><<<dim3(2, 1568), 256, 0, stream>>>(
      attn, projw_h, proj_b, x, nullptr, x1);
  // 5) LN2 -> bf16
  ln_kernel<false><<<25088, 256, 0, stream>>>(x1, n2g, n2b, h2);
  // 6) FC1 GEMM + bias + exact GELU -> bf16
  gemm_kernel<1024, 256, EPI_FC1><<<dim3(8, 1568), 256, 0, stream>>>(
      h2, fc1w_h, fc1_b, nullptr, fc1o, nullptr);
  // 7) FC2 GEMM + bias + x1 residual -> f32 output
  gemm_kernel<256, 1024, EPI_FC2><<<dim3(2, 1568), 256, 0, stream>>>(
      fc1o, fc2w_h, fc2_b, x1, nullptr, (float*)d_out);
}
